// TreensformerBlock_BranchMLP_84533546320170
// MI455X (gfx1250) — matrix operand order repaired and checked by reading, not compiled, not run on hardware
//
#include <hip/hip_runtime.h>
#include <hip/hip_bf16.h>
#include <cstdint>

typedef __attribute__((ext_vector_type(16))) __bf16 v16bf;
typedef __attribute__((ext_vector_type(8)))  float  v8f;
typedef __attribute__((ext_vector_type(4)))  unsigned int u32x4;
typedef __attribute__((ext_vector_type(4)))  float  f32x4;
typedef __attribute__((address_space(3))) char lds_char_t;

static constexpr int B_   = 64;     // batch
static constexpr int NN   = 341;    // tree nodes
static constexpr int E_   = 512;
static constexpr int H_   = 8;
static constexpr int HD   = 64;
static constexpr int R_   = B_ * NN;   // 21824 rows (divisible by 64)
static constexpr int NPAD = 352;       // 22*16 padded seq length

__device__ __forceinline__ v8f wmma_bf16(v16bf a, v16bf b, v8f c) {
  // D = A(16x32 bf16) * B(32x16 bf16) + C(16x16 f32)
  return __builtin_amdgcn_wmma_f32_16x16x32_bf16(false, a, false, b, (short)0, c, false, false);
}

union Frag { v16bf v; u32x4 q[2]; };
union H8   { u32x4 q; __bf16 h[8]; };

// LDS byte offset of a generic pointer into a __shared__ object
__device__ __forceinline__ unsigned lds_off(const void* p) {
  return (unsigned)(size_t)(lds_char_t*)p;
}

// 16B global -> LDS async copy (ASYNCcnt-tracked Tensor/async path, gfx1250)
__device__ __forceinline__ void async_copy_b128(const __bf16* g, unsigned lds) {
  asm volatile("global_load_async_to_lds_b128 %0, %1, off"
               :: "v"(lds), "v"((unsigned long long)(size_t)g) : "memory");
}
__device__ __forceinline__ void wait_async0() {
  asm volatile("s_wait_asynccnt 0x0" ::: "memory");
}

// ---------------- weight conversion ----------------
__global__ void k_convert(const float* __restrict__ src, __bf16* __restrict__ dst, int n) {
  int i = blockIdx.x * blockDim.x + threadIdx.x;
  int stride = gridDim.x * blockDim.x;
  for (; i < n; i += stride) dst[i] = (__bf16)src[i];
}

// mlp_M: [25][2048][512] (j,e) -> transposed bf16 [25][512][2048] (e,j)
__global__ void k_convert_M(const float* __restrict__ src, __bf16* __restrict__ dst) {
  const int total = 25 * 2048 * 512;
  int i = blockIdx.x * blockDim.x + threadIdx.x;
  int stride = gridDim.x * blockDim.x;
  for (; i < total; i += stride) {
    int mat = i / (2048 * 512);
    int rem = i - mat * (2048 * 512);
    int j = rem >> 9;
    int e = rem & 511;
    dst[((size_t)mat * 512 + e) * 2048 + j] = (__bf16)src[i];
  }
}

// ---------------- layernorm (f32 in -> bf16 out) ----------------
__global__ void k_layernorm(const float* __restrict__ x, const float* __restrict__ g,
                            const float* __restrict__ b, __bf16* __restrict__ out) {
  __shared__ float rs[128];
  __shared__ float rq[128];
  const int row = blockIdx.x;
  const int t = threadIdx.x;            // 128 threads, 4 elems each
  const float* xr = x + (size_t)row * E_;
  f32x4 v = ((const f32x4*)xr)[t];
  float s = v[0] + v[1] + v[2] + v[3];
  float q = v[0]*v[0] + v[1]*v[1] + v[2]*v[2] + v[3]*v[3];
  rs[t] = s; rq[t] = q;
  __syncthreads();
  for (int off = 64; off > 0; off >>= 1) {
    if (t < off) { rs[t] += rs[t + off]; rq[t] += rq[t + off]; }
    __syncthreads();
  }
  const float mean = rs[0] * (1.0f / E_);
  const float var  = rq[0] * (1.0f / E_) - mean * mean;
  const float rstd = rsqrtf(var + 1e-5f);
  f32x4 gv = ((const f32x4*)g)[t];
  f32x4 bv = ((const f32x4*)b)[t];
  __bf16* o = out + (size_t)row * E_ + t * 4;
  #pragma unroll
  for (int i = 0; i < 4; ++i) o[i] = (__bf16)((v[i] - mean) * rstd * gv[i] + bv[i]);
}

// ---------------- generic bf16 GEMM: C = A(MxK) @ B(KxN) + bias (+resid) ----------------
// 256 threads = 8 waves; block tile 64x128; each wave: 16-row strip x 64 cols = 4 WMMA tiles.
// A staged with async global->LDS; B loaded+transposed into LDS; next-B prefetched.
template<bool OUT_BF16, bool RES>
__global__ __launch_bounds__(256)
void k_gemm(const __bf16* __restrict__ A, const __bf16* __restrict__ Bm,
            const float* __restrict__ bias, const float* __restrict__ resid,
            void* __restrict__ Cout, int K, int lda, int ldb, int ldc) {
  __shared__ __align__(16) __bf16 As[64][40];    // [m][k], pad 8
  __shared__ __align__(16) __bf16 Bt[128][40];   // [n][k], pad 8 (B transposed in LDS)
  const int t = threadIdx.x;
  const int lane = t & 31, wave = t >> 5;
  const int wm = wave & 3, wn = wave >> 2;       // 4 row strips x 2 col groups of 64
  const int row0 = blockIdx.y * 64, col0 = blockIdx.x * 128;
  const int m = lane & 15, hi = lane >> 4;
  const int ar = t >> 2, acg = (t & 3) * 8;      // A staging: 64 rows x 4 groups of 8
  const int brow = t >> 3, bcg = (t & 7) * 16;   // B staging: 32 rows x 8 groups of 16

  const unsigned as_dst = lds_off(&As[ar][acg]);
  v8f acc0{}, acc1{}, acc2{}, acc3{};
  for (int kb = 0; kb < K; kb += 32) {
    async_copy_b128(A + (size_t)(row0 + ar) * lda + kb + acg, as_dst);
    const __bf16* bp = Bm + (size_t)(kb + brow) * ldb + col0 + bcg;
    if (kb + 32 < K) __builtin_prefetch(bp + (size_t)32 * ldb, 0, 1);
    H8 b0, b1;
    b0.q = *(const u32x4*)bp;
    b1.q = *(const u32x4*)(bp + 8);
    #pragma unroll
    for (int c = 0; c < 8; ++c) {
      Bt[bcg + c][brow]     = b0.h[c];
      Bt[bcg + 8 + c][brow] = b1.h[c];
    }
    wait_async0();
    __syncthreads();
    Frag fa;
    fa.q[0] = *(const u32x4*)&As[wm*16 + m][hi*8];
    fa.q[1] = *(const u32x4*)&As[wm*16 + m][16 + hi*8];
    #pragma unroll
    for (int tn = 0; tn < 4; ++tn) {
      const int nl = wn * 64 + tn * 16 + m;
      Frag fb;
      fb.q[0] = *(const u32x4*)&Bt[nl][hi*8];
      fb.q[1] = *(const u32x4*)&Bt[nl][16 + hi*8];
      v8f& acc = (tn == 0) ? acc0 : (tn == 1) ? acc1 : (tn == 2) ? acc2 : acc3;
      acc = wmma_bf16(fa.v, fb.v, acc);
    }
    __syncthreads();
  }
  #pragma unroll
  for (int tn = 0; tn < 4; ++tn) {
    v8f& acc = (tn == 0) ? acc0 : (tn == 1) ? acc1 : (tn == 2) ? acc2 : acc3;
    const int col = col0 + wn * 64 + tn * 16 + m;
    const float bv = bias[col];
    #pragma unroll
    for (int j = 0; j < 8; ++j) {
      const int row = row0 + wm * 16 + j + 8 * hi;
      float val = acc[j] + bv;
      if (RES) val += resid[(size_t)row * ldc + col];
      if (OUT_BF16) ((__bf16*)Cout)[(size_t)row * ldc + col] = (__bf16)val;
      else          ((float*)Cout)[(size_t)row * ldc + col] = val;
    }
  }
}

// ---------------- fused attention: one wave per (b, h, 16-row q tile) ----------------
__global__ __launch_bounds__(32)
void k_attn(const __bf16* __restrict__ QKV, __bf16* __restrict__ O) {
  __shared__ __align__(16) float  S[16][NPAD];    // scores f32
  __shared__ __align__(16) __bf16 P[16][NPAD];    // unnormalized probs bf16
  __shared__ __align__(16) __bf16 Vt[16][32];     // V chunk transposed [n][k]
  __shared__ float red[32];
  __shared__ float rsum[16];
  const int lane = threadIdx.x;
  const int qt = blockIdx.x, h = blockIdx.y, b = blockIdx.z;
  const int m = lane & 15, hi = lane >> 4;

  // Q fragments (A layout): rows m, features 0..63 in two K=32 chunks
  const int qrow = qt * 16 + m;
  const int gq = b * NN + (qrow < NN ? qrow : NN - 1);
  const __bf16* qp = QKV + (size_t)gq * 1536 + h * HD;
  Frag aq0, aq1;
  aq0.q[0] = *(const u32x4*)(qp + hi*8);
  aq0.q[1] = *(const u32x4*)(qp + 16 + hi*8);
  aq1.q[0] = *(const u32x4*)(qp + 32 + hi*8);
  aq1.q[1] = *(const u32x4*)(qp + 48 + hi*8);

  // scores = Q @ K^T / 8 ; K row-major -> B-fragment lanes read 16B runs at k = 8*hi and 16 + 8*hi
  for (int kt = 0; kt < 22; ++kt) {
    const int krow = kt * 16 + m;                 // m serves as fragment col n
    const int gk = b * NN + (krow < NN ? krow : NN - 1);
    const __bf16* kp = QKV + (size_t)gk * 1536 + 512 + h * HD;
    Frag bk0, bk1;
    bk0.q[0] = *(const u32x4*)(kp + hi*8);
    bk0.q[1] = *(const u32x4*)(kp + 16 + hi*8);
    bk1.q[0] = *(const u32x4*)(kp + 32 + hi*8);
    bk1.q[1] = *(const u32x4*)(kp + 48 + hi*8);
    v8f acc{};
    acc = wmma_bf16(aq0.v, bk0.v, acc);
    acc = wmma_bf16(aq1.v, bk1.v, acc);
    const int ncol = kt * 16 + m;
    const bool valid = ncol < NN;
    #pragma unroll
    for (int j = 0; j < 8; ++j)
      S[j + 8*hi][ncol] = valid ? acc[j] * 0.125f : -3.0e38f;
  }
  __syncthreads();

  // softmax: 2 lanes per row, 176 cols each; keep P unnormalized, scale at writeback
  float mx = -3.0e38f;
  for (int c = hi * 176; c < hi * 176 + 176; ++c) mx = fmaxf(mx, S[m][c]);
  red[lane] = mx;
  __syncthreads();
  const float rowmax = fmaxf(red[m], red[m + 16]);
  __syncthreads();
  float sm = 0.f;
  for (int c = hi * 176; c < hi * 176 + 176; ++c) {
    float e = __expf(S[m][c] - rowmax);
    P[m][c] = (__bf16)e;
    sm += e;
  }
  red[lane] = sm;
  __syncthreads();
  if (lane < 16) rsum[lane] = red[lane] + red[lane + 16];
  __syncthreads();

  // O = P @ V : 4 column tiles of 16, K over 352 in 32-chunks; V transposed via LDS
  for (int ct = 0; ct < 4; ++ct) {
    v8f acc{};
    for (int kc = 0; kc < 11; ++kc) {
      const int vrow = kc * 32 + lane;
      const int gv = b * NN + (vrow < NN ? vrow : NN - 1);
      const __bf16* vp = QKV + (size_t)gv * 1536 + 1024 + h * HD + ct * 16;
      H8 v0, v1;
      v0.q = *(const u32x4*)(vp);
      v1.q = *(const u32x4*)(vp + 8);
      __syncthreads();
      #pragma unroll
      for (int c = 0; c < 8; ++c) { Vt[c][lane] = v0.h[c]; Vt[c + 8][lane] = v1.h[c]; }
      __syncthreads();
      Frag ap, bv;
      ap.q[0] = *(const u32x4*)&P[m][kc*32 + hi*8];
      ap.q[1] = *(const u32x4*)&P[m][kc*32 + 16 + hi*8];
      bv.q[0] = *(const u32x4*)&Vt[m][hi*8];
      bv.q[1] = *(const u32x4*)&Vt[m][16 + hi*8];
      acc = wmma_bf16(ap.v, bv.v, acc);
    }
    #pragma unroll
    for (int j = 0; j < 8; ++j) {
      const int rloc = j + 8 * hi;
      const int grow = qt * 16 + rloc;
      if (grow < NN) {
        float val = acc[j] / rsum[rloc];
        O[((size_t)(b * NN + grow)) * 512 + h * HD + ct * 16 + m] = (__bf16)val;
      }
    }
  }
}

// ---------------- branch MLP: per-node chain GEMM, 64(batch) x 2048, K=(d+1)*512 ----------------
__global__ __launch_bounds__(256)
void k_branch(const __bf16* __restrict__ X, const __bf16* __restrict__ Mt,
              const float* __restrict__ b1, __bf16* __restrict__ Hout) {
  __shared__ __align__(16) __bf16 As[64][40];
  __shared__ __align__(16) __bf16 Bt[128][40];
  const int offs[6] = {0, 1, 5, 21, 85, 341};
  const int node = blockIdx.y;
  const int col0 = blockIdx.x * 128;
  const int d = (node >= 85) ? 4 : (node >= 21) ? 3 : (node >= 5) ? 2 : (node >= 1) ? 1 : 0;
  int chain[5];
  chain[0] = node;
  {
    int cur = node;
    for (int k = 1; k <= d; ++k) {
      const int dd = d - k + 1;                        // depth of cur
      cur = offs[dd - 1] + ((cur - offs[dd]) >> 2);    // parent
      chain[k] = cur;
    }
  }
  const int K = (d + 1) * 512;
  const int t = threadIdx.x;
  const int lane = t & 31, wave = t >> 5;
  const int wm = wave & 3, wn = wave >> 2;
  const int m = lane & 15, hi = lane >> 4;
  const int ar = t >> 2, acg = (t & 3) * 8;
  const int brow = t >> 3, bcg = (t & 7) * 16;

  const unsigned as_dst = lds_off(&As[ar][acg]);
  v8f acc0{}, acc1{}, acc2{}, acc3{};
  for (int kb = 0; kb < K; kb += 32) {
    const int kterm = kb >> 9;          // which ancestor
    const int e0 = kb & 511;            // feature base within term
    const int g = ar * NN + chain[kterm];
    async_copy_b128(X + (size_t)g * E_ + e0 + acg, as_dst);
    const int bi = d - kterm;           // matrix index M[d][bi]
    const size_t browg = (size_t)(d * 5 + bi) * 512 + (e0 + brow);
    const __bf16* bp = Mt + browg * 2048 + col0 + bcg;
    if (kb + 32 < K) __builtin_prefetch(bp + (size_t)32 * 2048, 0, 1);
    H8 b0, b1;
    b0.q = *(const u32x4*)bp;
    b1.q = *(const u32x4*)(bp + 8);
    #pragma unroll
    for (int c = 0; c < 8; ++c) {
      Bt[bcg + c][brow]     = b0.h[c];
      Bt[bcg + 8 + c][brow] = b1.h[c];
    }
    wait_async0();
    __syncthreads();
    Frag fa;
    fa.q[0] = *(const u32x4*)&As[wm*16 + m][hi*8];
    fa.q[1] = *(const u32x4*)&As[wm*16 + m][16 + hi*8];
    #pragma unroll
    for (int tn = 0; tn < 4; ++tn) {
      const int nl = wn * 64 + tn * 16 + m;
      Frag fb;
      fb.q[0] = *(const u32x4*)&Bt[nl][hi*8];
      fb.q[1] = *(const u32x4*)&Bt[nl][16 + hi*8];
      v8f& acc = (tn == 0) ? acc0 : (tn == 1) ? acc1 : (tn == 2) ? acc2 : acc3;
      acc = wmma_bf16(fa.v, fb.v, acc);
    }
    __syncthreads();
  }
  #pragma unroll
  for (int tn = 0; tn < 4; ++tn) {
    v8f& acc = (tn == 0) ? acc0 : (tn == 1) ? acc1 : (tn == 2) ? acc2 : acc3;
    const int col = col0 + wn * 64 + tn * 16 + m;
    const float bv = b1[(size_t)node * 2048 + col];
    #pragma unroll
    for (int j = 0; j < 8; ++j) {
      const int bb = wm * 16 + j + 8 * hi;           // batch index 0..63
      float val = fmaxf(acc[j] + bv, 0.f);
      Hout[((size_t)bb * NN + node) * 2048 + col] = (__bf16)val;
    }
  }
}

// ---------------- host launch ----------------
extern "C" void kernel_launch(void* const* d_in, const int* in_sizes, int n_in,
                              void* d_out, int out_size, void* d_ws, size_t ws_size,
                              hipStream_t stream) {
  (void)in_sizes; (void)n_in; (void)out_size; (void)ws_size;
  const float* x      = (const float*)d_in[0];
  const float* ln1_g  = (const float*)d_in[1];
  const float* ln1_b  = (const float*)d_in[2];
  const float* w_qkv  = (const float*)d_in[3];
  const float* b_qkv  = (const float*)d_in[4];
  const float* w_proj = (const float*)d_in[5];
  const float* b_proj = (const float*)d_in[6];
  const float* ln2_g  = (const float*)d_in[7];
  const float* ln2_b  = (const float*)d_in[8];
  const float* mlp_M  = (const float*)d_in[9];
  const float* mlp_b1 = (const float*)d_in[10];
  const float* w2     = (const float*)d_in[11];
  const float* b2     = (const float*)d_in[12];
  float* out = (float*)d_out;
  char* ws = (char*)d_ws;

  // workspace layout (bytes, 256-aligned); h aliases dead qkv+att region
  __bf16* wqkv_b  = (__bf16*)(ws + 0);           //  1.5 MB
  __bf16* wproj_b = (__bf16*)(ws + 1572864);     //  0.5 MB
  __bf16* w2_b    = (__bf16*)(ws + 2097152);     //  2.0 MB
  __bf16* mt_b    = (__bf16*)(ws + 4194304);     // 50.0 MB (transposed mlp_M)
  __bf16* ln_b    = (__bf16*)(ws + 56623104);    // 21.3 MB (LN1 then LN2)
  __bf16* qkv_b   = (__bf16*)(ws + 78970880);    // 63.9 MB
  __bf16* att_b   = (__bf16*)(ws + 146014208);   // 21.3 MB
  __bf16* h_b     = (__bf16*)(ws + 78970880);    // 85.2 MB (reuses qkv+att)

  // weights -> bf16 (mlp_M transposed)
  k_convert  <<<dim3(1024), 256, 0, stream>>>(w_qkv,  wqkv_b,  512 * 1536);
  k_convert  <<<dim3(512),  256, 0, stream>>>(w_proj, wproj_b, 512 * 512);
  k_convert  <<<dim3(1024), 256, 0, stream>>>(w2,     w2_b,    2048 * 512);
  k_convert_M<<<dim3(8192), 256, 0, stream>>>(mlp_M,  mt_b);

  // phase 1: attention
  k_layernorm<<<dim3(R_), 128, 0, stream>>>(x, ln1_g, ln1_b, ln_b);
  k_gemm<true, false><<<dim3(1536 / 128, R_ / 64), 256, 0, stream>>>(
      ln_b, wqkv_b, b_qkv, nullptr, qkv_b, 512, 512, 1536, 1536);
  k_attn<<<dim3(22, H_, B_), 32, 0, stream>>>(qkv_b, att_b);
  k_gemm<false, true><<<dim3(512 / 128, R_ / 64), 256, 0, stream>>>(
      att_b, wproj_b, b_proj, x, out, 512, 512, 512, 512);

  // phase 2: branch MLP
  k_layernorm<<<dim3(R_), 128, 0, stream>>>(out, ln2_g, ln2_b, ln_b);
  k_branch<<<dim3(2048 / 128, NN), 256, 0, stream>>>(ln_b, mt_b, mlp_b1, h_b);
  k_gemm<false, true><<<dim3(512 / 128, R_ / 64), 256, 0, stream>>>(
      h_b, w2_b, b2, out, out, 2048, 2048, 512, 512);
}
